// CaptioningRNN_60292750901482
// MI455X (gfx1250) — matrix operand order repaired and checked by reading, not compiled, not run on hardware
//
#include <hip/hip_runtime.h>
#include <math.h>

typedef __attribute__((ext_vector_type(16))) _Float16 v16h;
typedef __attribute__((ext_vector_type(8)))  _Float16 v8h;
typedef __attribute__((ext_vector_type(8)))  float    v8f;

// Problem dims (fixed by reference)
#define NB    64      // batch
#define TT    32      // timesteps
#define DA    1280    // CNN channels
#define PP    16      // spatial positions (4x4)
#define WD    300     // word dim
#define WPAD  320     // padded to mult of 32
#define HD    512     // hidden
#define H4    2048    // 4*H
#define VV    32000   // vocab
#define NT    2048    // N*T rows

// ---------------------------------------------------------------------------
// Pack kernels: build f16 operands (A row-major [MxK], B transposed [NcxK])
// ---------------------------------------------------------------------------
__global__ void pack_f32_to_f16(const float* __restrict__ src, _Float16* __restrict__ dst, int n) {
    int i = blockIdx.x * blockDim.x + threadIdx.x;
    if (i < n) dst[i] = (_Float16)src[i];
}

// AimgT[col=n*16+p][c] = A[n][c][p]   -> [1024 x 1280]
__global__ void pack_aimgT(const float* __restrict__ A, _Float16* __restrict__ dst) {
    int i = blockIdx.x * blockDim.x + threadIdx.x;
    if (i >= 1024 * DA) return;
    int col = i / DA, c = i - col * DA;
    int n = col >> 4, p = col & 15;
    dst[i] = (_Float16)A[(size_t)n * DA * PP + c * PP + p];
}

// WxT[j][k] = (k<300)? Wx[k][j] : 0    -> [2048 x 320]
__global__ void pack_wxT(const float* __restrict__ Wx, _Float16* __restrict__ dst) {
    int i = blockIdx.x * blockDim.x + threadIdx.x;
    if (i >= H4 * WPAD) return;
    int j = i / WPAD, k = i - j * WPAD;
    dst[i] = (k < WD) ? (_Float16)Wx[(size_t)k * H4 + j] : (_Float16)0.0f;
}

// Xmat[r=t*64+n][k] = (k<300)? W_embed[captions[n][t]][k] : 0  -> [2048 x 320]
__global__ void pack_xmat(const float* __restrict__ Wemb, const int* __restrict__ cap,
                          _Float16* __restrict__ dst) {
    int i = blockIdx.x * blockDim.x + threadIdx.x;
    if (i >= NT * WPAD) return;
    int r = i / WPAD, k = i - r * WPAD;
    int t = r >> 6, n = r & 63;
    if (k < WD) {
        int c = cap[n * TT + t];
        dst[i] = (_Float16)Wemb[(size_t)c * WD + k];
    } else {
        dst[i] = (_Float16)0.0f;
    }
}

// WcatT[j][k] = k<512 ? Wh[k][j] : Wattn[k-512][j]   -> [2048 x 1024]
__global__ void pack_wcatT(const float* __restrict__ Wh, const float* __restrict__ Wattn,
                           _Float16* __restrict__ dst) {
    int i = blockIdx.x * blockDim.x + threadIdx.x;
    if (i >= H4 * 1024) return;
    int j = i / 1024, k = i - j * 1024;
    float v = (k < HD) ? Wh[(size_t)k * H4 + j] : Wattn[(size_t)(k - HD) * H4 + j];
    dst[i] = (_Float16)v;
}

// WvT[v][k] = W_vocab[k][v]   -> [32000 x 512]
__global__ void pack_wvT(const float* __restrict__ Wv, _Float16* __restrict__ dst) {
    int i = blockIdx.x * blockDim.x + threadIdx.x;
    if (i >= VV * HD) return;
    int v = i / HD, k = i - v * HD;
    dst[i] = (_Float16)Wv[(size_t)k * VV + v];
}

// ---------------------------------------------------------------------------
// WMMA fragment loaders (per ISA 7.12.2 layouts, f16, wave32)
// A 16x32 frag: lane(m=l16,s): halves K=[k0+8s,+8) and [k0+16+8s,+8)
// B 32x16 frag: lane(n=l16,s): halves K=[k0+8s,+8) and [k0+16+8s,+8)
// ---------------------------------------------------------------------------
__device__ __forceinline__ v16h load_afrag(const _Float16* __restrict__ rowp, int k0, int s) {
    v8h lo = *(const v8h*)(rowp + k0 + 8 * s);
    v8h hi = *(const v8h*)(rowp + k0 + 16 + 8 * s);
    return __builtin_shufflevector(lo, hi, 0,1,2,3,4,5,6,7,8,9,10,11,12,13,14,15);
}
__device__ __forceinline__ v16h load_bfrag(const _Float16* __restrict__ rowp, int k0, int s) {
    v8h lo = *(const v8h*)(rowp + k0 + 8 * s);
    v8h hi = *(const v8h*)(rowp + k0 + 16 + 8 * s);
    return __builtin_shufflevector(lo, hi, 0,1,2,3,4,5,6,7,8,9,10,11,12,13,14,15);
}

// ---------------------------------------------------------------------------
// Register-blocked WMMA GEMM: each wave computes a 32x64 super-tile
// (2 M-frags x 4 N-frags -> 8 wmma per K=32 step, 12 b128 loads).
// C[M x Nc] = A[M x K] * BT[Nc x K]^T (+bias)(+addend). K % 32 == 0,
// M % 32 == 0, Nc % 64 == 0. bias_mode: 0 none, 1 per-col, 2 per-row.
// remap: vocab output row remap (t*64+n -> n*32+t).
// ---------------------------------------------------------------------------
__global__ __launch_bounds__(128)
void wmma_gemm_blk(const _Float16* __restrict__ A, const _Float16* __restrict__ BT,
                   float* __restrict__ C, const float* __restrict__ bias, int bias_mode,
                   const float* __restrict__ addend, int M, int K, int Nc, int remap) {
    const int wave = threadIdx.x >> 5;
    const int lane = threadIdx.x & 31;
    const int sid  = blockIdx.x * 4 + wave;   // super-tile id
    const int nsup = Nc >> 6;                 // super-tiles along N
    const int ms = sid / nsup;
    const int ns = sid - ms * nsup;
    if (ms * 32 >= M) return;                 // wave-uniform; EXEC stays all-ones

    const int l16 = lane & 15;
    const int s   = lane >> 4;

    const _Float16* arow0 = A + (size_t)(ms * 32 + l16) * K;
    const _Float16* arow1 = A + (size_t)(ms * 32 + 16 + l16) * K;
    const _Float16* brow0 = BT + (size_t)(ns * 64 +  0 + l16) * K;
    const _Float16* brow1 = BT + (size_t)(ns * 64 + 16 + l16) * K;
    const _Float16* brow2 = BT + (size_t)(ns * 64 + 32 + l16) * K;
    const _Float16* brow3 = BT + (size_t)(ns * 64 + 48 + l16) * K;

    v8f acc00 = {}, acc01 = {}, acc02 = {}, acc03 = {};
    v8f acc10 = {}, acc11 = {}, acc12 = {}, acc13 = {};

    for (int k0 = 0; k0 < K; k0 += 32) {
        if (k0 + 128 < K) {   // speculative prefetch of next B chunks (256B ahead)
            __builtin_prefetch(brow0 + k0 + 128, 0, 1);
            __builtin_prefetch(brow1 + k0 + 128, 0, 1);
            __builtin_prefetch(brow2 + k0 + 128, 0, 1);
            __builtin_prefetch(brow3 + k0 + 128, 0, 1);
        }
        v16h a0 = load_afrag(arow0, k0, s);
        v16h a1 = load_afrag(arow1, k0, s);
        v16h b0 = load_bfrag(brow0, k0, s);
        v16h b1 = load_bfrag(brow1, k0, s);
        v16h b2 = load_bfrag(brow2, k0, s);
        v16h b3 = load_bfrag(brow3, k0, s);
        acc00 = __builtin_amdgcn_wmma_f32_16x16x32_f16(false, a0, false, b0, (short)0, acc00, false, false);
        acc01 = __builtin_amdgcn_wmma_f32_16x16x32_f16(false, a0, false, b1, (short)0, acc01, false, false);
        acc02 = __builtin_amdgcn_wmma_f32_16x16x32_f16(false, a0, false, b2, (short)0, acc02, false, false);
        acc03 = __builtin_amdgcn_wmma_f32_16x16x32_f16(false, a0, false, b3, (short)0, acc03, false, false);
        acc10 = __builtin_amdgcn_wmma_f32_16x16x32_f16(false, a1, false, b0, (short)0, acc10, false, false);
        acc11 = __builtin_amdgcn_wmma_f32_16x16x32_f16(false, a1, false, b1, (short)0, acc11, false, false);
        acc12 = __builtin_amdgcn_wmma_f32_16x16x32_f16(false, a1, false, b2, (short)0, acc12, false, false);
        acc13 = __builtin_amdgcn_wmma_f32_16x16x32_f16(false, a1, false, b3, (short)0, acc13, false, false);
    }

    v8f* accs[2][4] = { {&acc00,&acc01,&acc02,&acc03}, {&acc10,&acc11,&acc12,&acc13} };
#pragma unroll
    for (int mi = 0; mi < 2; ++mi) {
#pragma unroll
        for (int j = 0; j < 4; ++j) {
            const v8f& a = *accs[mi][j];
            const int col = ns * 64 + j * 16 + l16;
#pragma unroll
            for (int r = 0; r < 8; ++r) {
                int row = ms * 32 + mi * 16 + r + 8 * s;  // C/D: VGPR r, lanes16-31 -> M=8+r
                float v = a[r];
                if (bias_mode == 1)      v += bias[col];
                else if (bias_mode == 2) v += bias[row];
                if (addend) v += addend[(size_t)row * Nc + col];
                int orow = remap ? ((row & 63) * TT + (row >> 6)) : row;
                C[(size_t)orow * Nc + col] = v;
            }
        }
    }
}

// ---------------------------------------------------------------------------
// 1-tile-per-wave WMMA GEMM (max parallelism; used for the small latency-
// bound recurrent GEMM, M=64). Same operand layouts and options.
// ---------------------------------------------------------------------------
__global__ __launch_bounds__(128)
void wmma_gemm_small(const _Float16* __restrict__ A, const _Float16* __restrict__ BT,
                     float* __restrict__ C, const float* __restrict__ bias, int bias_mode,
                     const float* __restrict__ addend, int M, int K, int Nc, int remap) {
    const int wave = threadIdx.x >> 5;
    const int lane = threadIdx.x & 31;
    const int tid  = blockIdx.x * 4 + wave;
    const int ntiles = Nc >> 4;
    const int mt = tid / ntiles;
    const int nt = tid - mt * ntiles;
    if (mt * 16 >= M) return;

    const int l16 = lane & 15;
    const int s   = lane >> 4;
    const _Float16* arow = A  + (size_t)(mt * 16 + l16) * K;
    const _Float16* brow = BT + (size_t)(nt * 16 + l16) * K;

    v8f acc = {};
    for (int k0 = 0; k0 < K; k0 += 32) {
        v16h af = load_afrag(arow, k0, s);
        v16h bf = load_bfrag(brow, k0, s);
        acc = __builtin_amdgcn_wmma_f32_16x16x32_f16(false, af, false, bf,
                                                     (short)0, acc, false, false);
    }
    const int col = nt * 16 + l16;
#pragma unroll
    for (int r = 0; r < 8; ++r) {
        int row = mt * 16 + r + 8 * s;
        float v = acc[r];
        if (bias_mode == 1)      v += bias[col];
        else if (bias_mode == 2) v += bias[row];
        if (addend) v += addend[(size_t)row * Nc + col];
        int orow = remap ? ((row & 63) * TT + (row >> 6)) : row;
        C[(size_t)orow * Nc + col] = v;
    }
}

// ---------------------------------------------------------------------------
// h0 = c0 = mean over P of A_proj (P1 layout [o][n*16+p], bias already inside)
// ---------------------------------------------------------------------------
__global__ void h0_kernel(const float* __restrict__ P1, float* __restrict__ h,
                          float* __restrict__ c) {
    int idx = blockIdx.x * blockDim.x + threadIdx.x;
    if (idx >= NB * HD) return;
    int n = idx >> 9, o = idx & 511;
    float s = 0.f;
#pragma unroll
    for (int p = 0; p < PP; ++p) s += P1[(size_t)o * 1024 + n * PP + p];
    s *= (1.0f / 16.0f);
    h[idx] = s;
    c[idx] = s;
}

// ---------------------------------------------------------------------------
// Per-step attention: scores -> softmax -> attn; emit HA16 = f16([h | attn])
// ---------------------------------------------------------------------------
__global__ __launch_bounds__(256)
void attn_kernel(const float* __restrict__ P1, const float* __restrict__ h,
                 _Float16* __restrict__ HA16) {
    const int n = blockIdx.x;
    __shared__ float sc[16][16];
    __shared__ float w[16];
    const int tid = threadIdx.x;
    const int p = tid & 15, chunk = tid >> 4;

    float partial = 0.f;
    const int obase = chunk * 32;
    for (int o = obase; o < obase + 32; ++o)
        partial += h[n * HD + o] * P1[(size_t)o * 1024 + n * PP + p];
    sc[p][chunk] = partial;
    __syncthreads();

    if (tid < 16) {
        float s = 0.f;
        for (int cch = 0; cch < 16; ++cch) s += sc[tid][cch];
        sc[tid][0] = s * 0.04419417382415922f;   // 1/sqrt(512)
    }
    __syncthreads();
    if (tid == 0) {
        float mx = sc[0][0];
        for (int i = 1; i < 16; ++i) mx = fmaxf(mx, sc[i][0]);
        float e[16], ssum = 0.f;
        for (int i = 0; i < 16; ++i) { e[i] = __expf(sc[i][0] - mx); ssum += e[i]; }
        float inv = 1.0f / ssum;
        for (int i = 0; i < 16; ++i) w[i] = e[i] * inv;
    }
    __syncthreads();

    for (int o = tid; o < HD; o += 256) {
        float av = 0.f;
#pragma unroll
        for (int pp2 = 0; pp2 < PP; ++pp2)
            av += P1[(size_t)o * 1024 + n * PP + pp2] * w[pp2];
        HA16[n * 1024 + HD + o] = (_Float16)av;
        HA16[n * 1024 + o]      = (_Float16)h[n * HD + o];
    }
}

// ---------------------------------------------------------------------------
// LSTM pointwise: gates, state update, emit f16 hidden row into Hseq16
// ---------------------------------------------------------------------------
__device__ __forceinline__ float sigmoidf_(float x) { return 1.0f / (1.0f + __expf(-x)); }

__global__ void lstm_pointwise(const float* __restrict__ a, float* __restrict__ h,
                               float* __restrict__ c, _Float16* __restrict__ Hseq16, int t) {
    int idx = blockIdx.x * blockDim.x + threadIdx.x;
    if (idx >= NB * HD) return;
    int n = idx >> 9, o = idx & 511;
    const float* ar = a + (size_t)n * H4;
    float gi = sigmoidf_(ar[o]);
    float gf = sigmoidf_(ar[HD + o]);
    float go = sigmoidf_(ar[2 * HD + o]);
    float gg = tanhf(ar[3 * HD + o]);
    float cn = gf * c[idx] + gi * gg;
    float hn = go * tanhf(cn);
    c[idx] = cn;
    h[idx] = hn;
    Hseq16[((size_t)t * NB + n) * HD + o] = (_Float16)hn;
}

// ---------------------------------------------------------------------------
// Launch
// ---------------------------------------------------------------------------
extern "C" void kernel_launch(void* const* d_in, const int* in_sizes, int n_in,
                              void* d_out, int out_size, void* d_ws, size_t ws_size,
                              hipStream_t stream) {
    const float* A       = (const float*)d_in[0];
    const int*   caps    = (const int*)  d_in[1];
    const float* conv_w  = (const float*)d_in[2];
    const float* conv_b  = (const float*)d_in[3];
    const float* Wx      = (const float*)d_in[4];
    const float* Wh      = (const float*)d_in[5];
    const float* Wattn   = (const float*)d_in[6];
    const float* bb      = (const float*)d_in[7];
    const float* Wemb    = (const float*)d_in[8];
    const float* Wvoc    = (const float*)d_in[9];
    const float* b_voc   = (const float*)d_in[10];
    float* out = (float*)d_out;

    // workspace carve (256B aligned)
    char* ws = (char*)d_ws;
    size_t off = 0;
    auto carve = [&](size_t bytes) { void* p = ws + off; off = (off + bytes + 255) & ~(size_t)255; return p; };
    _Float16* WC16   = (_Float16*)carve((size_t)HD * DA * 2);       // [512 x 1280]
    _Float16* AIMGT  = (_Float16*)carve((size_t)1024 * DA * 2);     // [1024 x 1280]
    float*    P1     = (float*)   carve((size_t)HD * 1024 * 4);     // [512 x 1024]
    float*    HBUF   = (float*)   carve((size_t)NB * HD * 4);
    float*    CBUF   = (float*)   carve((size_t)NB * HD * 4);
    _Float16* XMAT   = (_Float16*)carve((size_t)NT * WPAD * 2);     // [2048 x 320]
    _Float16* WXT    = (_Float16*)carve((size_t)H4 * WPAD * 2);     // [2048 x 320]
    float*    XPRE   = (float*)   carve((size_t)NT * H4 * 4);       // [2048 x 2048]
    _Float16* WCATT  = (_Float16*)carve((size_t)H4 * 1024 * 2);     // [2048 x 1024]
    _Float16* HA16   = (_Float16*)carve((size_t)NB * 1024 * 2);     // [64 x 1024]
    float*    ABUF   = (float*)   carve((size_t)NB * H4 * 4);       // [64 x 2048]
    _Float16* HSEQ16 = (_Float16*)carve((size_t)NT * HD * 2);       // [2048 x 512]
    _Float16* WVT    = (_Float16*)carve((size_t)VV * HD * 2);       // [32000 x 512]
    (void)ws_size; (void)n_in; (void)in_sizes; (void)out_size;

    const int TPB = 256;
    auto blk = [](size_t n, int t) { return (unsigned)((n + t - 1) / t); };

    // ---- pack f16 operands ----
    pack_f32_to_f16<<<blk((size_t)HD * DA, TPB), TPB, 0, stream>>>(conv_w, WC16, HD * DA);
    pack_aimgT     <<<blk((size_t)1024 * DA, TPB), TPB, 0, stream>>>(A, AIMGT);
    pack_wxT       <<<blk((size_t)H4 * WPAD, TPB), TPB, 0, stream>>>(Wx, WXT);
    pack_xmat      <<<blk((size_t)NT * WPAD, TPB), TPB, 0, stream>>>(Wemb, caps, XMAT);
    pack_wcatT     <<<blk((size_t)H4 * 1024, TPB), TPB, 0, stream>>>(Wh, Wattn, WCATT);
    pack_wvT       <<<blk((size_t)VV * HD, TPB), TPB, 0, stream>>>(Wvoc, WVT);

    // ---- 1x1 conv projection: P1[o][n*16+p] = conv_w @ AimgT^T + conv_b[o] ----
    // super-tiles: (512/32)*(1024/64) = 256 -> 64 blocks
    wmma_gemm_blk<<<(HD/32)*(1024/64)/4, 128, 0, stream>>>(WC16, AIMGT, P1, conv_b, /*row*/2,
                                                           nullptr, HD, DA, 1024, 0);
    // ---- h0 = c0 = spatial mean ----
    h0_kernel<<<blk((size_t)NB * HD, TPB), TPB, 0, stream>>>(P1, HBUF, CBUF);

    // ---- precompute XPRE[t*64+n][j] = x @ Wx + b ----
    // super-tiles: (2048/32)*(2048/64) = 2048 -> 512 blocks
    wmma_gemm_blk<<<(NT/32)*(H4/64)/4, 128, 0, stream>>>(XMAT, WXT, XPRE, bb, /*col*/1,
                                                         nullptr, NT, WPAD, H4, 0);

    // ---- recurrent timesteps (small GEMM -> 1 tile/wave for parallelism) ----
    for (int t = 0; t < TT; ++t) {
        attn_kernel<<<NB, 256, 0, stream>>>(P1, HBUF, HA16);
        wmma_gemm_small<<<(NB/16)*(H4/16)/4, 128, 0, stream>>>(HA16, WCATT, ABUF, nullptr, 0,
                                                               XPRE + (size_t)t * NB * H4,
                                                               NB, 1024, H4, 0);
        lstm_pointwise<<<blk((size_t)NB * HD, TPB), TPB, 0, stream>>>(ABUF, HBUF, CBUF, HSEQ16, t);
    }

    // ---- vocab projection with row remap (t*64+n -> n*32+t) + b_vocab ----
    // super-tiles: (2048/32)*(32000/64) = 32000 -> 8000 blocks
    wmma_gemm_blk<<<(NT/32)*(VV/64)/4, 128, 0, stream>>>(HSEQ16, WVT, out, b_voc, 1,
                                                         nullptr, NT, HD, VV, 1);
}
